// MobileMamba_blck_55327768708711
// MI455X (gfx1250) — matrix operand order repaired and checked by reading, not compiled, not run on hardware
//
#include <hip/hip_runtime.h>
#include <math.h>

// ---------------- types / helpers ----------------
typedef __attribute__((ext_vector_type(16))) __bf16 v16bf;
typedef __attribute__((ext_vector_type(8)))  float  v8f;

union ABFrag {
    v16bf v;
    unsigned short us[16];
    uint4 q[2];
};

__device__ __forceinline__ unsigned short f2bf(float f) {
    unsigned int u = __float_as_uint(f);
    unsigned int r = u + 0x7FFFu + ((u >> 16) & 1u);
    return (unsigned short)(r >> 16);
}
__device__ __forceinline__ float siluf(float x) { return x / (1.0f + expf(-x)); }
__device__ __forceinline__ float geluf(float x) { return 0.5f * x * (1.0f + erff(x * 0.70710678118654752f)); }
__device__ __forceinline__ float bnf(float x, float g, float b, float m, float v) {
    float s = g * rsqrtf(v + 1e-5f);
    return x * s + (b - m * s);
}

// ---------------- WMMA GEMM: C(MxN) = A(MxK,bf16,rowmajor) * BT(NxK,bf16,rowmajor) ----------------
// BT holds B transposed: BT[n][k] = B[k][n], so each lane's B fragment is 32 contiguous bytes.
// block = 128 threads = 4 waves; wave w: rows [by*64+w*16, +16), cols [bx*64, +64) as 4 16x16 tiles.
// K-loop is software-pipelined: next step's fragments are in flight while current WMMAs issue.
// M multiple of 16 (guarded per-wave), N multiple of 64, K multiple of 32, lda/ldb multiple of 8.
__global__ void wmma_gemm_bf16(const unsigned short* __restrict__ A,
                               const unsigned short* __restrict__ BT,
                               float* __restrict__ C,
                               int M, int N, int K,
                               int lda, int ldb, int ldc,
                               long long aStride, long long bStride, long long cStride) {
    const int z = blockIdx.z;
    A  += (size_t)z * aStride;
    BT += (size_t)z * bStride;
    C  += (size_t)z * cStride;

    const int tid  = threadIdx.x;
    const int wave = tid >> 5;
    const int lane = tid & 31;
    const int mrow = blockIdx.y * 64 + wave * 16;
    if (mrow >= M) return;                 // wave-uniform; EXEC stays all-1s for live waves
    const int nbase   = blockIdx.x * 64;
    const int halfsel = lane >> 4;         // 0: lanes 0-15, 1: lanes 16-31
    const int l15     = lane & 15;

    // A fragment: row = mrow+l15, K = k0+halfsel*16 .. +15 contiguous (matches B)
    const unsigned short* Arow = A + (size_t)(mrow + l15) * lda + halfsel * 16;
    // B fragment: col = nbase+t*16+l15, K = k0+halfsel*16 .. +15 contiguous in BT
    const unsigned short* Bcol[4];
#pragma unroll
    for (int t = 0; t < 4; ++t)
        Bcol[t] = BT + (size_t)(nbase + t * 16 + l15) * ldb + halfsel * 16;

    v8f acc[4] = {};

    // ---- prologue: load K-step 0 ----
    ABFrag af, bf[4];
    {
        const uint4* pa = (const uint4*)(Arow);
        af.q[0] = pa[0];
        af.q[1] = pa[1];
#pragma unroll
        for (int t = 0; t < 4; ++t) {
            const uint4* pb = (const uint4*)(Bcol[t]);
            bf[t].q[0] = pb[0];
            bf[t].q[1] = pb[1];
        }
    }

    // ---- pipelined main loop: prefetch k0, compute k0-32 ----
    for (int k0 = 32; k0 < K; k0 += 32) {
        ABFrag afn, bfn[4];
        const uint4* pa = (const uint4*)(Arow + k0);
        afn.q[0] = pa[0];
        afn.q[1] = pa[1];
#pragma unroll
        for (int t = 0; t < 4; ++t) {
            const uint4* pb = (const uint4*)(Bcol[t] + k0);
            bfn[t].q[0] = pb[0];
            bfn[t].q[1] = pb[1];
        }
#pragma unroll
        for (int t = 0; t < 4; ++t)
            acc[t] = __builtin_amdgcn_wmma_f32_16x16x32_bf16(
                false, af.v, false, bf[t].v, (short)0, acc[t], false, false);
        af = afn;
#pragma unroll
        for (int t = 0; t < 4; ++t) bf[t] = bfn[t];
    }

    // ---- epilogue: last K-step ----
#pragma unroll
    for (int t = 0; t < 4; ++t)
        acc[t] = __builtin_amdgcn_wmma_f32_16x16x32_bf16(
            false, af.v, false, bf[t].v, (short)0, acc[t], false, false);

#pragma unroll
    for (int t = 0; t < 4; ++t) {
        const int col = nbase + t * 16 + l15;
#pragma unroll
        for (int j = 0; j < 8; ++j) {
            const int row = mrow + halfsel * 8 + j;   // C/D layout: VGPR j -> M=j (+8 upper half)
            C[(size_t)row * ldc + col] = acc[t][j];
        }
    }
}

// ---------------- pack / convert kernels ----------------
__global__ void k_f32_to_bf16(const float* __restrict__ in, unsigned short* __restrict__ out, int n) {
    int i = blockIdx.x * blockDim.x + threadIdx.x;
    if (i < n) out[i] = f2bf(in[i]);
}
// rows r < Fvalid copied, rows Fvalid..Fpad zero.  in (Fvalid x K) -> out (Fpad x K)
__global__ void k_padRows(const float* __restrict__ in, unsigned short* __restrict__ out,
                          int Fvalid, int Fpad, int K) {
    int i = blockIdx.x * blockDim.x + threadIdx.x;
    if (i >= Fpad * K) return;
    int r = i / K;
    out[i] = (r < Fvalid) ? f2bf(in[i]) : (unsigned short)0;
}

// ---------------- conv1 im2col (BT layout): x(8,96,32,32) -> col(8, 1024, 864) bf16 ----------------
__global__ void k_im2col1(const float* __restrict__ x, unsigned short* __restrict__ col, int n) {
    int i = blockIdx.x * blockDim.x + threadIdx.x;
    if (i >= n) return;
    int k = i % 864, rest = i / 864;
    int p = rest & 1023, b = rest >> 10;
    int ci = k / 9, r9 = k % 9, kh = r9 / 3, kw = r9 % 3;
    int h = p >> 5, w = p & 31;
    int ih = h + kh - 1, iw = w + kw - 1;
    float v = 0.f;
    if ((unsigned)ih < 32u && (unsigned)iw < 32u)
        v = x[(((size_t)b * 96 + ci) * 32 + ih) * 32 + iw];
    col[i] = f2bf(v);
}

// ---------------- BN+GELU epilogue (conv1): c1f(8,96,1024) -> t0 bf16 BT layout (8,1024,96) ----------------
__global__ void k_ep1(const float* __restrict__ cf, unsigned short* __restrict__ out,
                      const float* g, const float* b, const float* m, const float* v, int n) {
    int i = blockIdx.x * blockDim.x + threadIdx.x;
    if (i >= n) return;
    int p = i & 1023, ch = (i >> 10) % 96, bb = i / (96 * 1024);
    float t = geluf(bnf(cf[i], g[ch], b[ch], m[ch], v[ch]));
    out[((size_t)bb * 1024 + p) * 96 + ch] = f2bf(t);
}

// ---------------- BN+SiLU epilogue (conv2) + NCHW->(B,L,D) transpose, bf16 ----------------
__global__ void k_ep2(const float* __restrict__ cf, unsigned short* __restrict__ xseq,
                      const float* g, const float* b, const float* m, const float* v, int n) {
    int i = blockIdx.x * blockDim.x + threadIdx.x;
    if (i >= n) return;
    int p = i & 1023, ch = (i >> 10) % 192, bb = i / (192 * 1024);
    float t = siluf(bnf(cf[i], g[ch], b[ch], m[ch], v[ch]));
    xseq[((size_t)bb * 1024 + p) * 192 + ch] = f2bf(t);
}

// ---------------- depthwise causal conv1d (K=4) + bias + SiLU ----------------
// xz(8192,768): xi = cols 0..383.   writes xc f32 and bf16, layout (8192,384)
__global__ void k_conv1d(const float* __restrict__ xz, const float* __restrict__ cw,
                         const float* __restrict__ cb, float* __restrict__ xcf,
                         unsigned short* __restrict__ xcb, int n) {
    int i = blockIdx.x * blockDim.x + threadIdx.x;
    if (i >= n) return;
    int ch = i % 384, l = (i / 384) & 1023, b = i / (384 * 1024);
    float acc = cb[ch];
#pragma unroll
    for (int k = 0; k < 4; ++k) {
        int ll = l - 3 + k;
        if (ll >= 0) acc += cw[ch * 4 + k] * xz[((size_t)b * 1024 + ll) * 768 + ch];
    }
    float t = siluf(acc);
    xcf[i] = t;
    xcb[i] = f2bf(t);
}

// ---------------- dt = softplus(dbl[:, :12] @ dt_w.T + dt_b) ----------------
__global__ void k_dt(const float* __restrict__ dbl, const float* __restrict__ dt_w,
                     const float* __restrict__ dt_b, float* __restrict__ dtf, int n) {
    int i = blockIdx.x * blockDim.x + threadIdx.x;
    if (i >= n) return;
    int d = i % 384;
    size_t idx = (size_t)(i / 384);
    float acc = dt_b[d];
#pragma unroll
    for (int r = 0; r < 12; ++r) acc += dbl[idx * 64 + r] * dt_w[d * 12 + r];
    dtf[i] = (acc > 20.f) ? acc : log1pf(expf(acc));
}

// ---------------- selective scan: one thread per (b,d), 16 states in registers ----------------
__global__ void k_scan(const float* __restrict__ dtf, const float* __restrict__ xcf,
                       const float* __restrict__ xzf, const float* __restrict__ dbl,
                       const float* __restrict__ A_log, const float* __restrict__ Dp,
                       unsigned short* __restrict__ yb) {
    int gid = blockIdx.x * blockDim.x + threadIdx.x;
    if (gid >= 8 * 384) return;
    int b = gid / 384, d = gid % 384;
    float a[16], h[16];
#pragma unroll
    for (int nn = 0; nn < 16; ++nn) { a[nn] = -expf(A_log[d * 16 + nn]); h[nn] = 0.f; }
    float dp = Dp[d];
    for (int l = 0; l < 1024; ++l) {
        size_t idx = (size_t)b * 1024 + l;
        size_t row = idx * 384 + d;
        float dt = dtf[row];
        float xv = xcf[row];
        float zv = xzf[idx * 768 + 384 + d];
        const float* db = dbl + idx * 64;
        float yv = 0.f;
#pragma unroll
        for (int nn = 0; nn < 16; ++nn) {
            float dA = expf(dt * a[nn]);
            h[nn] = dA * h[nn] + dt * db[12 + nn] * xv;
            yv += h[nn] * db[28 + nn];
        }
        float out = (yv + xv * dp) * siluf(zv);
        yb[row] = f2bf(out);
    }
}

// ---------------- im2col conv4 (BT layout, fused BN3+SiLU); src (b, p, o=192) -> col(8,1024,1728) ----------------
__global__ void k_im2col4(const float* __restrict__ c3f, unsigned short* __restrict__ col,
                          const float* g, const float* b, const float* m, const float* v, int n) {
    int i = blockIdx.x * blockDim.x + threadIdx.x;
    if (i >= n) return;
    int k = i % 1728, rest = i / 1728;
    int p = rest & 1023, bb = rest >> 10;
    int ci = k / 9, r9 = k % 9, kh = r9 / 3, kw = r9 % 3;
    int h = p >> 5, w = p & 31;
    int ih = h + kh - 1, iw = w + kw - 1;
    float t = 0.f;
    if ((unsigned)ih < 32u && (unsigned)iw < 32u) {
        float raw = c3f[((size_t)bb * 1024 + (ih * 32 + iw)) * 192 + ci];
        t = siluf(bnf(raw, g[ci], b[ci], m[ci], v[ci]));
    }
    col[i] = f2bf(t);
}

// ---------------- final BN4+GELU -> d_out (NCHW fp32, same flat layout as c4f) ----------------
__global__ void k_ep4(const float* __restrict__ cf, float* __restrict__ out,
                      const float* g, const float* b, const float* m, const float* v, int n) {
    int i = blockIdx.x * blockDim.x + threadIdx.x;
    if (i >= n) return;
    int ch = (i >> 10) % 192;
    out[i] = geluf(bnf(cf[i], g[ch], b[ch], m[ch], v[ch]));
}

// ---------------- host ----------------
#define EL(n) <<<(((n) + 255) / 256), 256, 0, stream>>>

extern "C" void kernel_launch(void* const* d_in, const int* in_sizes, int n_in,
                              void* d_out, int out_size, void* d_ws, size_t ws_size,
                              hipStream_t stream) {
    const float* x    = (const float*)d_in[0];
    const float* w1   = (const float*)d_in[1];
    const float *g1 = (const float*)d_in[2], *b1 = (const float*)d_in[3],
                *m1 = (const float*)d_in[4], *v1 = (const float*)d_in[5];
    const float* w2   = (const float*)d_in[6];
    const float *g2 = (const float*)d_in[7], *b2 = (const float*)d_in[8],
                *m2 = (const float*)d_in[9], *v2 = (const float*)d_in[10];
    const float* in_w = (const float*)d_in[11];
    const float* cw   = (const float*)d_in[12];
    const float* cb   = (const float*)d_in[13];
    const float* xp_w = (const float*)d_in[14];
    const float* dt_w = (const float*)d_in[15];
    const float* dt_b = (const float*)d_in[16];
    const float* A_log= (const float*)d_in[17];
    const float* Dp   = (const float*)d_in[18];
    const float* out_w= (const float*)d_in[19];
    const float* w3   = (const float*)d_in[20];
    const float *g3 = (const float*)d_in[21], *b3 = (const float*)d_in[22],
                *m3 = (const float*)d_in[23], *v3 = (const float*)d_in[24];
    const float* w4   = (const float*)d_in[25];
    const float *g4 = (const float*)d_in[26], *b4 = (const float*)d_in[27],
                *m4 = (const float*)d_in[28], *v4 = (const float*)d_in[29];

    char* ws = (char*)d_ws;
    size_t off = 0;
    auto alloc = [&](size_t bytes) { size_t r = off; off = (off + bytes + 255) & ~(size_t)255; return r; };

    unsigned short* w1b   = (unsigned short*)(ws + alloc((size_t)96  * 864  * 2));  // A conv1
    unsigned short* w2b   = (unsigned short*)(ws + alloc((size_t)192 * 96   * 2));  // A conv2
    unsigned short* w4b   = (unsigned short*)(ws + alloc((size_t)192 * 1728 * 2));  // A conv4
    unsigned short* inwb  = (unsigned short*)(ws + alloc((size_t)768 * 192  * 2));  // BT in_proj (768x192)
    unsigned short* xpwb  = (unsigned short*)(ws + alloc((size_t)64  * 384  * 2));  // BT x_proj (64x384, pad)
    unsigned short* outwb = (unsigned short*)(ws + alloc((size_t)192 * 384  * 2));  // BT out_proj (192x384)
    unsigned short* w3b   = (unsigned short*)(ws + alloc((size_t)192 * 192  * 2));  // BT conv3 (192x192)
    unsigned short* col1  = (unsigned short*)(ws + alloc((size_t)8 * 1024 * 864 * 2));
    float*          c1f   = (float*)(ws + alloc((size_t)8 * 96  * 1024 * 4));
    unsigned short* t0b   = (unsigned short*)(ws + alloc((size_t)8 * 1024 * 96 * 2)); // BT (b,p,ci)
    float*          c2f   = (float*)(ws + alloc((size_t)8 * 192 * 1024 * 4));
    unsigned short* xseqb = (unsigned short*)(ws + alloc((size_t)8192 * 192 * 2));
    float*          xzf   = (float*)(ws + alloc((size_t)8192 * 768 * 4));
    float*          xcf   = (float*)(ws + alloc((size_t)8192 * 384 * 4));
    unsigned short* xcb   = (unsigned short*)(ws + alloc((size_t)8192 * 384 * 2));
    float*          dblf  = (float*)(ws + alloc((size_t)8192 * 64 * 4));
    float*          dtf   = (float*)(ws + alloc((size_t)8192 * 384 * 4));
    unsigned short* yb    = (unsigned short*)(ws + alloc((size_t)8192 * 384 * 2));
    float*          mof   = (float*)(ws + alloc((size_t)8192 * 192 * 4));
    unsigned short* mob   = (unsigned short*)(ws + alloc((size_t)8192 * 192 * 2));
    float*          c3f   = (float*)(ws + alloc((size_t)8192 * 192 * 4));
    unsigned short* col4  = (unsigned short*)(ws + alloc((size_t)8 * 1024 * 1728 * 2));
    float*          c4f   = (float*)(ws + alloc((size_t)8 * 192 * 1024 * 4));
    (void)ws_size; (void)in_sizes; (void)n_in; (void)out_size;

    // ---- weight packing (BT matrices are already (N,K) row-major in the inputs) ----
    k_f32_to_bf16 EL(96 * 864)  (w1,    w1b,   96 * 864);
    k_f32_to_bf16 EL(192 * 96)  (w2,    w2b,   192 * 96);
    k_f32_to_bf16 EL(192 * 1728)(w4,    w4b,   192 * 1728);
    k_f32_to_bf16 EL(768 * 192) (in_w,  inwb,  768 * 192);
    k_padRows     EL(64 * 384)  (xp_w,  xpwb,  44, 64, 384);
    k_f32_to_bf16 EL(192 * 384) (out_w, outwb, 192 * 384);
    k_f32_to_bf16 EL(192 * 192) (w3,    w3b,   192 * 192);

    // ---- stage 1: conv1 3x3 as implicit GEMM, then BN+GELU ----
    k_im2col1 EL(8 * 1024 * 864)(x, col1, 8 * 1024 * 864);
    wmma_gemm_bf16<<<dim3(16, 2, 8), 128, 0, stream>>>(
        w1b, col1, c1f, 96, 1024, 864, 864, 864, 1024,
        0LL, (long long)1024 * 864, (long long)96 * 1024);
    k_ep1 EL(8 * 96 * 1024)(c1f, t0b, g1, b1, m1, v1, 8 * 96 * 1024);

    // ---- stage 2: conv2 1x1 GEMM, BN+SiLU, transpose to (B,L,D) ----
    wmma_gemm_bf16<<<dim3(16, 3, 8), 128, 0, stream>>>(
        w2b, t0b, c2f, 192, 1024, 96, 96, 96, 1024,
        0LL, (long long)1024 * 96, (long long)192 * 1024);
    k_ep2 EL(8 * 192 * 1024)(c2f, xseqb, g2, b2, m2, v2, 8 * 192 * 1024);

    // ---- mamba in_proj: (8192x192) x (192x768) ----
    wmma_gemm_bf16<<<dim3(12, 128, 1), 128, 0, stream>>>(
        xseqb, inwb, xzf, 8192, 768, 192, 192, 192, 768, 0LL, 0LL, 0LL);

    // ---- depthwise conv1d + SiLU ----
    k_conv1d EL(8192 * 384)(xzf, cw, cb, xcf, xcb, 8192 * 384);

    // ---- x_proj: (8192x384) x (384x64pad) ----
    wmma_gemm_bf16<<<dim3(1, 128, 1), 128, 0, stream>>>(
        xcb, xpwb, dblf, 8192, 64, 384, 384, 384, 64, 0LL, 0LL, 0LL);

    // ---- dt projection + softplus ----
    k_dt EL(8192 * 384)(dblf, dt_w, dt_b, dtf, 8192 * 384);

    // ---- selective scan + gate ----
    k_scan<<<dim3(24), 128, 0, stream>>>(dtf, xcf, xzf, dblf, A_log, Dp, yb);

    // ---- out_proj: (8192x384) x (384x192) ----
    wmma_gemm_bf16<<<dim3(3, 128, 1), 128, 0, stream>>>(
        yb, outwb, mof, 8192, 192, 384, 384, 384, 192, 0LL, 0LL, 0LL);

    // ---- conv3 1x1 directly in (b,p,o) layout: (8192x192) x (192x192) ----
    k_f32_to_bf16 EL(8192 * 192)(mof, mob, 8192 * 192);
    wmma_gemm_bf16<<<dim3(3, 128, 1), 128, 0, stream>>>(
        mob, w3b, c3f, 8192, 192, 192, 192, 192, 192, 0LL, 0LL, 0LL);

    // ---- conv4 3x3: im2col (fused BN3+SiLU) + GEMM + BN4+GELU ----
    k_im2col4 EL(8 * 1024 * 1728)(c3f, col4, g3, b3, m3, v3, 8 * 1024 * 1728);
    wmma_gemm_bf16<<<dim3(16, 3, 8), 128, 0, stream>>>(
        w4b, col4, c4f, 192, 1024, 1728, 1728, 1728, 1024,
        0LL, (long long)1024 * 1728, (long long)192 * 1024);
    k_ep4 EL(8 * 192 * 1024)(c4f, (float*)d_out, g4, b4, m4, v4, 8 * 192 * 1024);
}
